// LigerFusedLinearJSD_2164663517660
// MI455X (gfx1250) — compile-verified, matrix operand order checked by reading, not run on hardware
//
#include <hip/hip_runtime.h>
#include <math.h>

// Problem constants (from the reference): BT, H, HT, V
#define BT_DIM 2048
#define H_S    2048
#define H_T    4096
#define V_DIM  32000

// GEMM tiling
#define BM 128
#define BN 128
#define BK 32
#define LDS_STRIDE 40   // 32 bf16 + 8 pad -> 80B row stride, rows 16B-aligned

typedef __attribute__((ext_vector_type(16))) __bf16 v16bf;
typedef __attribute__((ext_vector_type(8)))  float  v8f;

union FragBF16 {
    v16bf v;
    uint4 q[2];
};

// Native f32 -> bf16 conversion (fptrunc lowers to v_cvt_pk_bf16_f32-class
// hardware conversion on gfx1250, RNE), packed two-per-dword.
__device__ __forceinline__ unsigned int pack_bf16x2(float lo, float hi) {
    union { __bf16 h[2]; unsigned int u; } cv;
    cv.h[0] = (__bf16)lo;
    cv.h[1] = (__bf16)hi;
    return cv.u;
}

// Load 16 consecutive f32 from global, convert to 16 bf16 packed in 8 dwords.
__device__ __forceinline__ void load_convert16(const float* __restrict__ g,
                                               unsigned int* p) {
#pragma unroll
    for (int i = 0; i < 4; ++i) {
        float4 f = *(const float4*)(g + i * 4);
        p[i * 2 + 0] = pack_bf16x2(f.x, f.y);
        p[i * 2 + 1] = pack_bf16x2(f.z, f.w);
    }
}

// Store 16 bf16 (8 dwords) to LDS as two b128 stores.
__device__ __forceinline__ void store_lds16(unsigned short* s,
                                            const unsigned int* p) {
    *(uint4*)(s + 0) = make_uint4(p[0], p[1], p[2], p[3]);
    *(uint4*)(s + 8) = make_uint4(p[4], p[5], p[6], p[7]);
}

// One BK=32 MMA step for this wave: 2 A-frags x 4 B-frags -> 8 WMMAs.
__device__ __forceinline__ void mma_step(const unsigned short* __restrict__ Abuf,
                                         const unsigned short* __restrict__ Bbuf,
                                         int wave_m, int wave_n, int rlo, int kh,
                                         v8f acc[2][4]) {
    FragBF16 a[2], b[4];
#pragma unroll
    for (int mi = 0; mi < 2; ++mi) {
        const unsigned short* p = Abuf + (wave_m * 32 + mi * 16 + rlo) * LDS_STRIDE;
        a[mi].q[0] = *(const uint4*)(p + kh * 8);        // K b0..b0+7
        a[mi].q[1] = *(const uint4*)(p + 16 + kh * 8);   // K 16+b0..+7
    }
#pragma unroll
    for (int ni = 0; ni < 4; ++ni) {
        const unsigned short* p = Bbuf + (wave_n * 64 + ni * 16 + rlo) * LDS_STRIDE;
        b[ni].q[0] = *(const uint4*)(p + kh * 8);
        b[ni].q[1] = *(const uint4*)(p + 16 + kh * 8);
    }
#pragma unroll
    for (int mi = 0; mi < 2; ++mi)
#pragma unroll
        for (int ni = 0; ni < 4; ++ni)
            acc[mi][ni] = __builtin_amdgcn_wmma_f32_16x16x32_bf16(
                false, a[mi].v, false, b[ni].v,
                (short)0, acc[mi][ni], false, false);
}

// out[m, n] = dot(X[m, :], W[n, :])   (X: MxK row-major, W: NxK row-major)
// 128x128 block tile, 8 wave32s (4x2), double-buffered LDS with register
// prefetch so global loads overlap the WMMA pipe.
__global__ __launch_bounds__(256)
void gemm_rowmajor_bt_kernel(const float* __restrict__ X,
                             const float* __restrict__ W,
                             float* __restrict__ out,
                             int M, int N, int K) {
    __shared__ unsigned short As[2][BM * LDS_STRIDE];
    __shared__ unsigned short Bs[2][BN * LDS_STRIDE];

    const int tid    = threadIdx.x;
    const int lane   = tid & 31;
    const int wid    = tid >> 5;        // 0..7
    const int wave_m = wid >> 1;        // 0..3 -> 32-row slab
    const int wave_n = wid & 1;         // 0..1 -> 64-col slab

    const int block_n = blockIdx.x * BN;
    const int block_m = blockIdx.y * BM;

    const int rlo = lane & 15;          // row within 16
    const int kh  = lane >> 4;          // K-half selector

    v8f acc[2][4];
#pragma unroll
    for (int mi = 0; mi < 2; ++mi)
#pragma unroll
        for (int ni = 0; ni < 4; ++ni)
#pragma unroll
            for (int r = 0; r < 8; ++r)
                acc[mi][ni][r] = 0.0f;

    // Staging: 256 threads cover a 128x32 f32 tile -> 16 floats each
    const int ld_row = tid >> 1;          // 0..127
    const int ld_col = (tid & 1) * 16;    // 0 or 16

    const float* gA = X + (size_t)(block_m + ld_row) * K + ld_col;
    const float* gB = W + (size_t)(block_n + ld_row) * K + ld_col;
    const int s_off = ld_row * LDS_STRIDE + ld_col;

    unsigned int pa[8], pb[8];

    // Prologue: fill buffer 0 with the k=0 tile
    load_convert16(gA, pa);
    load_convert16(gB, pb);
    store_lds16(&As[0][s_off], pa);
    store_lds16(&Bs[0][s_off], pb);
    __syncthreads();

    int cur = 0;
    for (int k0 = BK; k0 < K; k0 += BK) {
        // Issue next tile's global loads first (overlap with WMMAs below)
        load_convert16(gA + k0, pa);
        load_convert16(gB + k0, pb);

        mma_step(As[cur], Bs[cur], wave_m, wave_n, rlo, kh, acc);

        const int nxt = cur ^ 1;
        store_lds16(&As[nxt][s_off], pa);
        store_lds16(&Bs[nxt][s_off], pb);
        __syncthreads();
        cur = nxt;
    }
    mma_step(As[cur], Bs[cur], wave_m, wave_n, rlo, kh, acc);

    // C/D layout: VGPR r -> row (m_base + r + (lane>>4)*8), col (n_base + (lane&15))
    // 32-bit index math (BT*V < 2^31 elements) off a single 64-bit base.
    const unsigned int uN   = (unsigned int)N;
    const unsigned int colb = (unsigned int)(block_n + wave_n * 64 + rlo);
    const unsigned int rowb = (unsigned int)(block_m + wave_m * 32 + kh * 8);
#pragma unroll
    for (int mi = 0; mi < 2; ++mi) {
#pragma unroll
        for (int ni = 0; ni < 4; ++ni) {
            unsigned int idx = (rowb + mi * 16) * uN + colb + ni * 16;
#pragma unroll
            for (int r = 0; r < 8; ++r)
                out[idx + (unsigned int)r * uN] = acc[mi][ni][r];
        }
    }
}

// ---------------- JSD epilogue ----------------

__device__ __forceinline__ float block_reduce_max(float v, float* sm) {
    const int tid = threadIdx.x;
    sm[tid] = v;
    __syncthreads();
    for (int s = 128; s > 0; s >>= 1) {
        if (tid < s) sm[tid] = fmaxf(sm[tid], sm[tid + s]);
        __syncthreads();
    }
    float r = sm[0];
    __syncthreads();
    return r;
}

__device__ __forceinline__ float block_reduce_sum(float v, float* sm) {
    const int tid = threadIdx.x;
    sm[tid] = v;
    __syncthreads();
    for (int s = 128; s > 0; s >>= 1) {
        if (tid < s) sm[tid] = sm[tid] + sm[tid + s];
        __syncthreads();
    }
    float r = sm[0];
    __syncthreads();
    return r;
}

__global__ __launch_bounds__(256)
void jsd_rows_kernel(const float* __restrict__ S,
                     const float* __restrict__ T,
                     float* __restrict__ out, int V, float inv_bt) {
    __shared__ float sm[256];
    const int row = blockIdx.x;
    const int tid = threadIdx.x;
    const float* srow = S + (size_t)row * V;
    const float* trow = T + (size_t)row * V;

    // pass 1: row maxima
    float ms = -INFINITY, mt = -INFINITY;
    for (int i = tid; i < V; i += 256) {
        ms = fmaxf(ms, srow[i]);
        mt = fmaxf(mt, trow[i]);
    }
    ms = block_reduce_max(ms, sm);
    mt = block_reduce_max(mt, sm);

    // pass 2: sum of exponentials
    float ss = 0.0f, st = 0.0f;
    for (int i = tid; i < V; i += 256) {
        ss += __expf(srow[i] - ms);
        st += __expf(trow[i] - mt);
    }
    ss = block_reduce_sum(ss, sm);
    st = block_reduce_sum(st, sm);
    const float lzs = __logf(ss) + ms;   // log Z_student
    const float lzt = __logf(st) + mt;   // log Z_teacher

    // pass 3: generalized JSD, beta = 0.5
    float acc = 0.0f;
    for (int i = tid; i < V; i += 256) {
        float Xl = srow[i] - lzs;        // student log-prob
        float Yl = trow[i] - lzt;        // teacher log-prob
        float Q  = __expf(Xl);
        float P  = __expf(Yl);
        float Mv = 0.5f * (P + Q);
        float lM = __logf(Mv);
        acc += 0.5f * P * (Yl - lM) + 0.5f * Q * (Xl - lM);
    }
    acc = block_reduce_sum(acc, sm);
    if (tid == 0) atomicAdd(out, acc * inv_bt);
}

__global__ void zero_out_kernel(float* out) {
    if (threadIdx.x == 0 && blockIdx.x == 0) out[0] = 0.0f;
}

extern "C" void kernel_launch(void* const* d_in, const int* in_sizes, int n_in,
                              void* d_out, int out_size, void* d_ws, size_t ws_size,
                              hipStream_t stream) {
    (void)in_sizes; (void)n_in; (void)out_size; (void)ws_size;

    const float* Xs = (const float*)d_in[0];   // student_input  (BT, H)
    const float* Ws = (const float*)d_in[1];   // student_weight (V, H)
    const float* Xt = (const float*)d_in[2];   // teacher_input  (BT, HT)
    const float* Wt = (const float*)d_in[3];   // teacher_weight (V, HT)
    float* out = (float*)d_out;

    float* s_log = (float*)d_ws;                        // (BT, V) student logits
    float* t_log = s_log + (size_t)BT_DIM * V_DIM;      // (BT, V) teacher logits

    zero_out_kernel<<<1, 64, 0, stream>>>(out);

    dim3 grid(V_DIM / BN, BT_DIM / BM);   // (250, 16)
    gemm_rowmajor_bt_kernel<<<grid, 256, 0, stream>>>(Xs, Ws, s_log,
                                                      BT_DIM, V_DIM, H_S);
    gemm_rowmajor_bt_kernel<<<grid, 256, 0, stream>>>(Xt, Wt, t_log,
                                                      BT_DIM, V_DIM, H_T);

    jsd_rows_kernel<<<BT_DIM, 256, 0, stream>>>(s_log, t_log, out,
                                                V_DIM, 1.0f / (float)BT_DIM);
}
